// CaMIL_26663156974056
// MI455X (gfx1250) — matrix operand order repaired and checked by reading, not compiled, not run on hardware
//
#include <hip/hip_runtime.h>
#include <hip/hip_bf16.h>

// ---------------------------------------------------------------------------
// Problem constants
// ---------------------------------------------------------------------------
#define ROI   116
#define FEAT  6670            // ROI*(ROI-1)/2
#define BW    8192            // B*W = 64*128
#define NB    64              // batch
#define NW    128             // bags per batch element
#define DH    1024            // D == H == 1024
#define KP1   6688            // FEAT padded to multiple of 32 (209 * 32)
#define KT1   209
#define KP2   7712            // (FEAT + H) padded to multiple of 32 (241 * 32)
#define KT2   241
#define NTB   64              // 1024 / 16 n-tiles

typedef __attribute__((ext_vector_type(16))) __bf16   v16bf;
typedef __attribute__((ext_vector_type(8)))  float    v8f;
typedef unsigned int u32x4 __attribute__((ext_vector_type(4)));

union Frag { u32x4 u[2]; v16bf v; };

__device__ __forceinline__ unsigned pack2bf(float a, float b) {
    __bf16 x = (__bf16)a, y = (__bf16)b;
    unsigned short ux = __builtin_bit_cast(unsigned short, x);
    unsigned short uy = __builtin_bit_cast(unsigned short, y);
    return (unsigned)ux | ((unsigned)uy << 16);
}

// ---------------------------------------------------------------------------
// 1) triu_indices LUT: f -> i*ROI + j  (i < j)
// ---------------------------------------------------------------------------
__global__ void build_lut(int* __restrict__ lut) {
    int f = blockIdx.x * 256 + threadIdx.x;
    if (f >= FEAT) return;
    int i = 0, r = f, cnt = ROI - 1;
    while (r >= cnt) { r -= cnt; ++i; --cnt; }
    lut[f] = i * ROI + (i + 1 + r);
}

// ---------------------------------------------------------------------------
// 2) gather upper triangle of FC -> row-major bf16 A matrix [BW, KP1]
// ---------------------------------------------------------------------------
__global__ void extract_bag(const float* __restrict__ FCp, const int* __restrict__ lut,
                            __bf16* __restrict__ bagA) {
    long idx = (long)blockIdx.x * 256 + threadIdx.x;
    if (idx >= (long)BW * KP1) return;
    int bag = (int)(idx / KP1);
    int kk  = (int)(idx % KP1);
    float v = 0.f;
    if (kk < FEAT) v = FCp[(size_t)bag * (ROI * ROI) + lut[kk]];
    bagA[idx] = (__bf16)v;
}

// ---------------------------------------------------------------------------
// 3) pack a [Ksrc, 1024] f32 weight into WMMA B-fragment tile layout (bf16).
//    Tile t = kt*64+nt holds 256 dwords; lane L owns dwords [L*8 .. L*8+7],
//    dword v packs B[k=kt*32 + (L/16)*16 + 2v][n] , B[k+1][n].
// ---------------------------------------------------------------------------
__global__ void packB(const float* __restrict__ W, unsigned* __restrict__ out, int Ksrc) {
    int t = blockIdx.x;           // tile index
    int d = threadIdx.x;          // 0..255
    int kt = t >> 6, nt = t & 63;
    int lane = d >> 3, v = d & 7;
    int k = kt * 32 + (lane >> 4) * 16 + 2 * v;
    int n = nt * 16 + (lane & 15);
    float f0 = (k     < Ksrc) ? W[(size_t)k       * DH + n] : 0.f;
    float f1 = (k + 1 < Ksrc) ? W[(size_t)(k + 1) * DH + n] : 0.f;
    out[(size_t)t * 256 + d] = pack2bf(f0, f1);
}

// ---------------------------------------------------------------------------
// 4) Generic bf16 WMMA GEMM:  C[M,1024] = act(A[M,lda] @ Bpacked + bias)
//    block = 8 waves (2x4), block tile 64x256; wave -> 32x64 (2x4 wmma tiles)
//    act: 0 none, 1 tanh, 2 relu.  M must be a multiple of 64 in A storage.
// ---------------------------------------------------------------------------
__global__ __launch_bounds__(256, 2)
void gemm_wmma(const __bf16* __restrict__ A, const unsigned* __restrict__ Bp,
               float* __restrict__ C, const float* __restrict__ bias,
               int M, int lda, int Kt, int act) {
    const int lane = threadIdx.x & 31;
    const int wave = threadIdx.x >> 5;
    const int wr = wave >> 2;        // 0..1 : +32 rows
    const int wc = wave & 3;         // 0..3 : +64 cols
    const int mrow = lane & 15;
    const int half = lane >> 4;

    const int m0  = blockIdx.y * 64 + wr * 32;
    const int nt0 = blockIdx.x * 16 + wc * 4;     // 4 n-tiles per wave

    v8f acc[2][4] = {};

    const __bf16* a0 = A + (size_t)(m0 + mrow) * lda + half * 16;
    const __bf16* a1 = a0 + (size_t)16 * lda;

    // single per-wave B base; the wave's 4 n-tiles are contiguous in the
    // packed layout (tile stride = 64 u32x4 = 1024 B), so all 8 B-fragment
    // loads use immediate offsets off one address -> global_load_b128 clause.
    const size_t bstep = (size_t)NTB * 32 * 2;    // u32x4 elements per kt
    const u32x4* bbase = reinterpret_cast<const u32x4*>(Bp) + ((size_t)nt0 * 32 + lane) * 2;

    for (int kt = 0; kt < Kt; ++kt) {
        Frag a[2], b[4];
        const u32x4* pa0 = reinterpret_cast<const u32x4*>(a0 + kt * 32);
        a[0].u[0] = pa0[0]; a[0].u[1] = pa0[1];
        const u32x4* pa1 = reinterpret_cast<const u32x4*>(a1 + kt * 32);
        a[1].u[0] = pa1[0]; a[1].u[1] = pa1[1];

        const u32x4* pb = bbase + (size_t)kt * bstep;
#pragma unroll
        for (int j = 0; j < 4; ++j) {
            b[j].u[0] = pb[j * 64];
            b[j].u[1] = pb[j * 64 + 1];
        }

        if (kt + 1 < Kt) {
            __builtin_prefetch(a0 + (kt + 1) * 32, 0, 3);
            __builtin_prefetch(a1 + (kt + 1) * 32, 0, 3);
        }

#pragma unroll
        for (int i = 0; i < 2; ++i)
#pragma unroll
            for (int j = 0; j < 4; ++j)
                acc[i][j] = __builtin_amdgcn_wmma_f32_16x16x32_bf16(
                    false, a[i].v, false, b[j].v, (short)0, acc[i][j], false, false);
    }

#pragma unroll
    for (int j = 0; j < 4; ++j) {
        int n = blockIdx.x * 256 + wc * 64 + j * 16 + mrow;
        float bn_ = bias ? bias[n] : 0.f;
#pragma unroll
        for (int i = 0; i < 2; ++i) {
            int mb = m0 + i * 16 + half * 8;
#pragma unroll
            for (int v = 0; v < 8; ++v) {
                int m = mb + v;
                if (m < M) {
                    float x = acc[i][j][v] + bn_;
                    if (act == 1) x = tanhf(x);
                    else if (act == 2) x = fmaxf(x, 0.f);
                    C[(size_t)m * DH + n] = x;
                }
            }
        }
    }
}

// ---------------------------------------------------------------------------
// 5) a = h @ W2 + b2 ; softmax over W=128 per batch element
// ---------------------------------------------------------------------------
__global__ void att_softmax(const float* __restrict__ h, const float* __restrict__ W2,
                            const float* __restrict__ b2, float* __restrict__ att) {
    int b = blockIdx.x, w = threadIdx.x;          // 128 threads
    const float* row = h + (size_t)(b * NW + w) * DH;
    float s = 0.f;
    for (int d = 0; d < DH; ++d) s += row[d] * W2[d];
    s += b2[0];
    __shared__ float red[NW];
    red[w] = s; __syncthreads();
    for (int off = 64; off; off >>= 1) { if (w < off) red[w] = fmaxf(red[w], red[w + off]); __syncthreads(); }
    float mx = red[0]; __syncthreads();
    float e = __expf(s - mx);
    red[w] = e; __syncthreads();
    for (int off = 64; off; off >>= 1) { if (w < off) red[w] += red[w + off]; __syncthreads(); }
    att[b * NW + w] = e / red[0];
}

// ---------------------------------------------------------------------------
// 6) bag_z_up[b,f] = sum_w att[b,w] * FC[b,w,lut[f]]  (fp32 accuracy path)
// ---------------------------------------------------------------------------
__global__ void bag_z(const float* __restrict__ FCp, const int* __restrict__ lut,
                      const float* __restrict__ att, float* __restrict__ raw) {
    __shared__ float ash[NW];
    int b = blockIdx.y;
    if (threadIdx.x < NW) ash[threadIdx.x] = att[b * NW + threadIdx.x];
    __syncthreads();
    int f = blockIdx.x * 256 + threadIdx.x;
    if (f >= FEAT) return;
    const float* base = FCp + (size_t)b * NW * (ROI * ROI) + lut[f];
    float s = 0.f;
    for (int w = 0; w < NW; ++w) s += ash[w] * base[(size_t)w * (ROI * ROI)];
    raw[(size_t)b * FEAT + f] = s;
}

// ---------------------------------------------------------------------------
// 7) batchnorm over batch (64) -> fp32 normalized + bf16 padded GEMM-A copy
// ---------------------------------------------------------------------------
__global__ void bn_bagz(const float* __restrict__ raw, const float* __restrict__ g,
                        const float* __restrict__ bt, float* __restrict__ outn,
                        __bf16* __restrict__ outbf) {
    int f = blockIdx.x * 256 + threadIdx.x;
    if (f >= KP1) return;
    if (f >= FEAT) {
        for (int b = 0; b < NB; ++b) outbf[(size_t)b * KP1 + f] = (__bf16)0.f;
        return;
    }
    float mu = 0.f;
    for (int b = 0; b < NB; ++b) mu += raw[(size_t)b * FEAT + f];
    mu *= (1.f / NB);
    float var = 0.f;
    for (int b = 0; b < NB; ++b) { float d = raw[(size_t)b * FEAT + f] - mu; var += d * d; }
    var *= (1.f / NB);
    float inv = rsqrtf(var + 1e-5f);
    float gg = g[f], bb = bt[f];
    for (int b = 0; b < NB; ++b) {
        float x = (raw[(size_t)b * FEAT + f] - mu) * inv * gg + bb;
        outn[(size_t)b * FEAT + f] = x;
        outbf[(size_t)b * KP1 + f] = (__bf16)x;
    }
}

// ---------------------------------------------------------------------------
// 8) batchnorm of cluster_centers (2 rows) -> bf16 A [64, KP1], rows 2..63 = 0
// ---------------------------------------------------------------------------
__global__ void bn_cc(const float* __restrict__ cc, const float* __restrict__ g,
                      const float* __restrict__ bt, __bf16* __restrict__ outbf) {
    int f = blockIdx.x * 256 + threadIdx.x;
    if (f >= KP1) return;
    __bf16 z = (__bf16)0.f;
    if (f < FEAT) {
        float x0 = cc[f], x1 = cc[FEAT + f];
        float mu = 0.5f * (x0 + x1);
        float d0 = x0 - mu, d1 = x1 - mu;
        float inv = rsqrtf(0.5f * (d0 * d0 + d1 * d1) + 1e-5f);
        float gg = g[f], bb = bt[f];
        outbf[f]                = (__bf16)(d0 * inv * gg + bb);
        outbf[(size_t)KP1 + f]  = (__bf16)(d1 * inv * gg + bb);
    } else { outbf[f] = z; outbf[(size_t)KP1 + f] = z; }
    for (int r = 2; r < NB; ++r) outbf[(size_t)r * KP1 + f] = z;
}

// ---------------------------------------------------------------------------
// 9) 2-way cross attention + final_cross -> X[:, 0:1024] (bf16)
// ---------------------------------------------------------------------------
__global__ void cross_attn(const float* __restrict__ q, const float* __restrict__ km,
                           const float* __restrict__ vm, __bf16* __restrict__ Xbf) {
    int b = blockIdx.x, t = threadIdx.x;          // 128 threads
    const float* qr = q + (size_t)b * DH;
    float s0 = 0.f, s1 = 0.f;
    for (int d = t; d < DH; d += 128) { float qq = qr[d]; s0 += qq * km[d]; s1 += qq * km[DH + d]; }
    __shared__ float r0[128], r1[128];
    r0[t] = s0; r1[t] = s1; __syncthreads();
    for (int off = 64; off; off >>= 1) {
        if (t < off) { r0[t] += r0[t + off]; r1[t] += r1[t + off]; }
        __syncthreads();
    }
    float a0 = r0[0] * (1.f / 32.f), a1 = r1[0] * (1.f / 32.f);   // /sqrt(1024)
    float m = fmaxf(a0, a1);
    float e0 = __expf(a0 - m), e1 = __expf(a1 - m);
    float p0 = e0 / (e0 + e1), p1 = e1 / (e0 + e1);
    for (int n = t; n < DH; n += 128)
        Xbf[(size_t)b * KP2 + n] = (__bf16)(p0 * vm[n] + p1 * vm[DH + n]);
}

// ---------------------------------------------------------------------------
// 10) X[:, 1024:] = bag_z_up (bf16), zero pad to KP2
// ---------------------------------------------------------------------------
__global__ void fill_X(const float* __restrict__ bagz_n, __bf16* __restrict__ Xbf) {
    int b = blockIdx.y;
    int idx = blockIdx.x * 256 + threadIdx.x;
    int col = DH + idx;
    if (col >= KP2) return;
    float v = (idx < FEAT) ? bagz_n[(size_t)b * FEAT + idx] : 0.f;
    Xbf[(size_t)b * KP2 + col] = (__bf16)v;
}

// ---------------------------------------------------------------------------
// 11) head: sigmoid(z @ Wc2 + bc2) -> Y_prob, Y_hat
// ---------------------------------------------------------------------------
__global__ void head(const float* __restrict__ z, const float* __restrict__ Wc2,
                     const float* __restrict__ bc2, float* __restrict__ out) {
    int b = blockIdx.x, t = threadIdx.x;          // 128 threads
    const float* zr = z + (size_t)b * DH;
    float s = 0.f;
    for (int d = t; d < DH; d += 128) s += zr[d] * Wc2[d];
    __shared__ float r[128];
    r[t] = s; __syncthreads();
    for (int off = 64; off; off >>= 1) { if (t < off) r[t] += r[t + off]; __syncthreads(); }
    if (t == 0) {
        float p = 1.f / (1.f + __expf(-(r[0] + bc2[0])));
        out[b] = p;
        out[NB + b] = (p >= 0.5f) ? 1.f : 0.f;
    }
}

// ---------------------------------------------------------------------------
// host launcher
// ---------------------------------------------------------------------------
extern "C" void kernel_launch(void* const* d_in, const int* in_sizes, int n_in,
                              void* d_out, int out_size, void* d_ws, size_t ws_size,
                              hipStream_t stream) {
    (void)in_sizes; (void)n_in; (void)out_size; (void)ws_size;
    const float* FCp = (const float*)d_in[0];
    const float* cc  = (const float*)d_in[1];
    const float* W1  = (const float*)d_in[2];
    const float* b1  = (const float*)d_in[3];
    const float* W2  = (const float*)d_in[4];
    const float* b2  = (const float*)d_in[5];
    const float* bng = (const float*)d_in[6];
    const float* bnb = (const float*)d_in[7];
    const float* bn2g= (const float*)d_in[8];
    const float* bn2b= (const float*)d_in[9];
    const float* Wq  = (const float*)d_in[10];
    const float* bq  = (const float*)d_in[11];
    const float* Wk  = (const float*)d_in[12];
    const float* bk  = (const float*)d_in[13];
    const float* Wc1 = (const float*)d_in[14];
    const float* bc1 = (const float*)d_in[15];
    const float* Wc2 = (const float*)d_in[16];
    const float* bc2 = (const float*)d_in[17];
    float* out = (float*)d_out;

    char* p = (char*)d_ws;
    auto carve = [&](size_t bytes) -> char* {
        char* r = p; p += (bytes + 255) & ~(size_t)255; return r;
    };
    int*      lut      = (int*)     carve((size_t)FEAT * 4);
    __bf16*   bagA     = (__bf16*)  carve((size_t)BW * KP1 * 2);
    unsigned* W1p      = (unsigned*)carve((size_t)KT1 * NTB * 256 * 4);
    unsigned* Wqp      = (unsigned*)carve((size_t)KT1 * NTB * 256 * 4);
    unsigned* Wkp      = (unsigned*)carve((size_t)KT1 * NTB * 256 * 4);
    unsigned* Wc1p     = (unsigned*)carve((size_t)KT2 * NTB * 256 * 4);
    float*    h        = (float*)   carve((size_t)BW * DH * 4);
    float*    att      = (float*)   carve((size_t)BW * 4);
    float*    bagz_raw = (float*)   carve((size_t)NB * FEAT * 4);
    float*    bagz_n   = (float*)   carve((size_t)NB * FEAT * 4);
    __bf16*   bagz_bf  = (__bf16*)  carve((size_t)NB * KP1 * 2);
    __bf16*   cc_bf    = (__bf16*)  carve((size_t)NB * KP1 * 2);
    float*    qb       = (float*)   carve((size_t)NB * DH * 4);
    float*    km       = (float*)   carve((size_t)NB * DH * 4);
    float*    vm       = (float*)   carve((size_t)NB * DH * 4);
    __bf16*   Xbf      = (__bf16*)  carve((size_t)NB * KP2 * 2);
    float*    zb       = (float*)   carve((size_t)NB * DH * 4);

    // stage 1: gather + pack
    build_lut<<<(FEAT + 255) / 256, 256, 0, stream>>>(lut);
    extract_bag<<<(int)(((long)BW * KP1 + 255) / 256), 256, 0, stream>>>(FCp, lut, bagA);
    packB<<<KT1 * NTB, 256, 0, stream>>>(W1,  W1p,  FEAT);
    packB<<<KT1 * NTB, 256, 0, stream>>>(Wq,  Wqp,  FEAT);
    packB<<<KT1 * NTB, 256, 0, stream>>>(Wk,  Wkp,  FEAT);
    packB<<<KT2 * NTB, 256, 0, stream>>>(Wc1, Wc1p, FEAT + DH);

    // stage 2: the big WMMA GEMM + attention pooling
    gemm_wmma<<<dim3(DH / 256, BW / 64), 256, 0, stream>>>(bagA, W1p, h, b1, BW, KP1, KT1, /*tanh*/1);
    att_softmax<<<NB, NW, 0, stream>>>(h, W2, b2, att);
    bag_z<<<dim3((FEAT + 255) / 256, NB), 256, 0, stream>>>(FCp, lut, att, bagz_raw);

    // stage 3: batchnorms
    bn_bagz<<<(KP1 + 255) / 256, 256, 0, stream>>>(bagz_raw, bng, bnb, bagz_n, bagz_bf);
    bn_cc<<<(KP1 + 255) / 256, 256, 0, stream>>>(cc, bn2g, bn2b, cc_bf);

    // stage 4: q / k / v projections (WMMA)
    gemm_wmma<<<dim3(DH / 256, 1), 256, 0, stream>>>(bagz_bf, Wqp, qb, bq, NB, KP1, KT1, 0);
    gemm_wmma<<<dim3(DH / 256, 1), 256, 0, stream>>>(cc_bf,   Wkp, km, bk, NB, KP1, KT1, 0);
    gemm_wmma<<<dim3(DH / 256, 1), 256, 0, stream>>>(cc_bf,   Wqp, vm, bq, NB, KP1, KT1, 0);

    // stage 5: cross attention + classifier head
    cross_attn<<<NB, 128, 0, stream>>>(qb, km, vm, Xbf);
    fill_X<<<dim3((KP2 - DH + 255) / 256, NB), 256, 0, stream>>>(bagz_n, Xbf);
    gemm_wmma<<<dim3(DH / 256, 1), 256, 0, stream>>>(Xbf, Wc1p, zb, bc1, NB, KP2, KT2, /*relu*/2);
    head<<<NB, 128, 0, stream>>>(zb, Wc2, bc2, out);
}
